// TransformerBlock_57793079935646
// MI455X (gfx1250) — compile-verified, matrix operand order checked by reading, not run on hardware
//
#include <hip/hip_runtime.h>
#include <hip/hip_bf16.h>

typedef __bf16 bf16;
typedef __attribute__((ext_vector_type(16))) __bf16 v16bf;
typedef __attribute__((ext_vector_type(8)))  float  v8f;

#define WMMA_BF16(a, b, c) \
  __builtin_amdgcn_wmma_f32_16x16x32_bf16(false, (a), false, (b), (short)0, (c), false, false)

constexpr int Cc   = 384;
constexpr int NHh  = 12;
constexpr int HD   = 32;
constexpr int N_   = 49;     // 7*7 tokens per window
constexpr int NW   = 1024;   // 16 images * 64 windows
constexpr int MTOK = NW * N_;                // 50176
constexpr float SCALE_ = 0.17677669529663687f; // 32^-0.5

// ---------------------------------------------------------------------------
// Weight convert + transpose: in fp32 [K, Nc] -> out bf16 [Nc, K]
// ---------------------------------------------------------------------------
__global__ __launch_bounds__(256)
void convert_transpose(const float* __restrict__ in, bf16* __restrict__ out,
                       int Kd, int Nc) {
  int idx = blockIdx.x * 256 + threadIdx.x;
  if (idx >= Kd * Nc) return;
  int k = idx / Nc, n = idx - k * Nc;
  out[(size_t)n * Kd + k] = (bf16)in[idx];
}

// ---------------------------------------------------------------------------
// LayerNorm (+ optional cyclic shift & window partition), fp32 in -> bf16 out
// One wave per token row of 384.
// ---------------------------------------------------------------------------
template<bool SHIFTED>
__global__ __launch_bounds__(256)
void ln_kernel(const float* __restrict__ x, const float* __restrict__ g,
               const float* __restrict__ beta, bf16* __restrict__ out) {
  const int lane = threadIdx.x & 31;
  const int wid  = threadIdx.x >> 5;
  const int t    = blockIdx.x * 8 + wid;          // window-space token id
  const float* src;
  if (SHIFTED) {
    int w = t / N_, n = t - w * N_;
    int b = w >> 6, wh = (w & 63) >> 3, ww2 = w & 7;
    int hs = wh * 7 + n / 7, ws2 = ww2 * 7 + n % 7;
    int i  = (hs + 3) % 56;        // forward roll(-3) on H
    int jj = (ws2 + 53) % 56;      // forward roll(+3) on W
    src = x + ((size_t)(b * 3136 + i * 56 + jj)) * Cc;
  } else {
    src = x + (size_t)t * Cc;
  }
  float v[12], s = 0.f, ss = 0.f;
  for (int u = 0; u < 12; ++u) {
    v[u] = src[lane + 32 * u];
    s += v[u]; ss += v[u] * v[u];
  }
  for (int off = 16; off > 0; off >>= 1) {
    s  += __shfl_xor(s,  off, 32);
    ss += __shfl_xor(ss, off, 32);
  }
  float mu  = s  * (1.f / 384.f);
  float var = ss * (1.f / 384.f) - mu * mu;
  float inv = rsqrtf(var + 1e-3f);
  bf16* dst = out + (size_t)t * Cc;
  for (int u = 0; u < 12; ++u) {
    int c = lane + 32 * u;
    dst[c] = (bf16)((v[u] - mu) * inv * g[c] + beta[c]);
  }
}

// ---------------------------------------------------------------------------
// Generic bf16 WMMA GEMM:  C[M,Nc] = A[M,K] * Bt[Nc,K]^T + bias
// One wave -> 16x64 tile. block = 128 (4 waves), grid = (Nc/64, M/64).
// MODE 0: QKV scatter (q scaled, v transposed into padded [.,HD,64])
// MODE 1: GELU(erf) -> bf16
// MODE 2: window-reverse + un-shift + residual -> fp32
// MODE 3: + residual -> fp32
// ---------------------------------------------------------------------------
template<int MODE>
__global__ __launch_bounds__(128)
void gemm_bf16_kernel(const bf16* __restrict__ A, const bf16* __restrict__ Bt,
                      const float* __restrict__ bias, int K, int Nc,
                      bf16* __restrict__ outb, float* __restrict__ outf,
                      const float* __restrict__ skip,
                      bf16* __restrict__ qb, bf16* __restrict__ kb,
                      bf16* __restrict__ vtb) {
  const int lane = threadIdx.x & 31;
  const int wid  = threadIdx.x >> 5;
  const int m0   = (blockIdx.y * 4 + wid) * 16;
  const int n0   = blockIdx.x * 64;
  const int l16  = lane & 15;
  const int kg   = (lane >> 4) << 4;

  const bf16* ap = A  + (size_t)(m0 + l16) * K + kg;
  const bf16* bp = Bt + (size_t)(n0 + l16) * K + kg;
  const size_t bs = (size_t)16 * K;

  v8f acc[4] = {};
  for (int k0 = 0; k0 < K; k0 += 32) {
    v16bf a  = *(const v16bf*)(ap + k0);
    v16bf b0 = *(const v16bf*)(bp + k0);
    v16bf b1 = *(const v16bf*)(bp + bs + k0);
    v16bf b2 = *(const v16bf*)(bp + 2 * bs + k0);
    v16bf b3 = *(const v16bf*)(bp + 3 * bs + k0);
    acc[0] = WMMA_BF16(a, b0, acc[0]);
    acc[1] = WMMA_BF16(a, b1, acc[1]);
    acc[2] = WMMA_BF16(a, b2, acc[2]);
    acc[3] = WMMA_BF16(a, b3, acc[3]);
  }

  const int mb = m0 + ((lane >> 4) << 3);
  for (int j = 0; j < 4; ++j) {
    const int n = n0 + j * 16 + l16;
    const float bn = bias[n];
    for (int r = 0; r < 8; ++r) {
      const int m = mb + r;
      float val = acc[j][r] + bn;
      if (MODE == 0) {
        int which = n / Cc;
        int c = n - which * Cc;
        int h = c >> 5, d = c & 31;
        int w = m / N_, nt = m - w * N_;
        if (which == 0)
          qb[(((size_t)(w * NHh + h)) * N_ + nt) * HD + d] = (bf16)(val * SCALE_);
        else if (which == 1)
          kb[(((size_t)(w * NHh + h)) * N_ + nt) * HD + d] = (bf16)val;
        else
          vtb[(((size_t)(w * NHh + h)) * HD + d) * 64 + nt] = (bf16)val;
      } else if (MODE == 1) {
        float ge = 0.5f * val * (1.0f + erff(val * 0.70710678118654752f));
        outb[(size_t)m * Nc + n] = (bf16)ge;
      } else if (MODE == 2) {
        int w = m / N_, nt = m - w * N_;
        int b = w >> 6, wh = (w & 63) >> 3, ww2 = w & 7;
        int hs = wh * 7 + nt / 7, ws2 = ww2 * 7 + nt % 7;
        int i  = (hs + 3) % 56;      // reverse roll(+3) on H
        int jj = (ws2 + 3) % 56;     // reverse roll(+3) on W (faithful)
        size_t idx = ((size_t)(b * 3136 + i * 56 + jj)) * Cc + n;
        outf[idx] = val + skip[idx];
      } else {
        size_t idx = (size_t)m * Nc + n;
        outf[idx] = val + skip[idx];
      }
    }
  }
}

// ---------------------------------------------------------------------------
// Windowed attention: one block (128 threads = 4 waves) per (window, head).
// S = q k^T via WMMA -> LDS; softmax (+rel-pos bias, +shift mask) scalar;
// P (bf16, LDS) x V^T via WMMA -> out [nW, N, C].
// ---------------------------------------------------------------------------
__global__ __launch_bounds__(128)
void attn_kernel(const bf16* __restrict__ q, const bf16* __restrict__ k,
                 const bf16* __restrict__ vt, const float* __restrict__ relt,
                 bf16* __restrict__ out) {
  __shared__ float Sl[64 * 64];
  __shared__ bf16  Pl[64 * 64];

  const int blk = blockIdx.x;
  const int w = blk / NHh, h = blk - w * NHh;
  const int lane = threadIdx.x & 31;
  const int wid  = threadIdx.x >> 5;
  const int l16  = lane & 15;
  const int kg   = (lane >> 4) << 4;
  const int m0   = wid * 16;

  const size_t qkbase = ((size_t)(w * NHh + h)) * N_ * HD;

  // ---- S = q k^T ----
  int rA = m0 + l16; if (rA > 48) rA = 48;           // clamp padded rows
  v16bf qa = *(const v16bf*)(q + qkbase + (size_t)rA * HD + kg);
  for (int nt = 0; nt < 4; ++nt) {
    int col = nt * 16 + l16; if (col > 48) col = 48;
    v16bf kb = *(const v16bf*)(k + qkbase + (size_t)col * HD + kg);
    v8f acc = {};
    acc = WMMA_BF16(qa, kb, acc);
    for (int r = 0; r < 8; ++r)
      Sl[(m0 + r + ((lane >> 4) << 3)) * 64 + nt * 16 + l16] = acc[r];
  }
  __syncthreads();

  // ---- softmax with rel-pos bias + shift mask ----
  const int m = threadIdx.x;
  if (m < 64) {
    if (m >= 49) {
      for (int n = 0; n < 64; ++n) Pl[m * 64 + n] = (bf16)0.f;
    } else {
      int wm = w & 63, wh = wm >> 3, ww2 = wm & 7;
      int mi = m / 7, mj = m - mi * 7;
      int hm = wh * 7 + mi, wmj = ww2 * 7 + mj;
      int regm = (hm < 49 ? 0 : (hm < 53 ? 1 : 2)) * 3 +
                 (wmj < 49 ? 0 : (wmj < 53 ? 1 : 2));
      float mx = -1e30f;
      for (int n = 0; n < 49; ++n) {
        int ni = n / 7, nj = n - ni * 7;
        int rel = (mi - ni + 6) * 13 + (mj - nj + 6);
        float bias = relt[rel * NHh + h];
        int hn = wh * 7 + ni, wnj = ww2 * 7 + nj;
        int regn = (hn < 49 ? 0 : (hn < 53 ? 1 : 2)) * 3 +
                   (wnj < 49 ? 0 : (wnj < 53 ? 1 : 2));
        float s = Sl[m * 64 + n] + bias + (regm != regn ? -100.f : 0.f);
        Sl[m * 64 + n] = s;
        mx = fmaxf(mx, s);
      }
      float sum = 0.f;
      for (int n = 0; n < 49; ++n) {
        float e = __expf(Sl[m * 64 + n] - mx);
        Sl[m * 64 + n] = e;
        sum += e;
      }
      float inv = 1.f / sum;
      for (int n = 0; n < 49; ++n) Pl[m * 64 + n] = (bf16)(Sl[m * 64 + n] * inv);
      for (int n = 49; n < 64; ++n) Pl[m * 64 + n] = (bf16)0.f;
    }
  }
  __syncthreads();

  // ---- out = P V ----
  const size_t vbase = ((size_t)(w * NHh + h)) * HD * 64;
  v8f o0 = {}, o1 = {};
  for (int k0 = 0; k0 < 64; k0 += 32) {
    v16bf pa  = *(const v16bf*)(&Pl[(m0 + l16) * 64 + k0 + kg]);
    v16bf vb0 = *(const v16bf*)(vt + vbase + (size_t)(l16) * 64 + k0 + kg);
    v16bf vb1 = *(const v16bf*)(vt + vbase + (size_t)(16 + l16) * 64 + k0 + kg);
    o0 = WMMA_BF16(pa, vb0, o0);
    o1 = WMMA_BF16(pa, vb1, o1);
  }
  for (int r = 0; r < 8; ++r) {
    int mm = m0 + r + ((lane >> 4) << 3);
    if (mm < 49) {
      size_t ob = ((size_t)(w * N_ + mm)) * Cc + h * HD;
      out[ob + l16]      = (bf16)o0[r];
      out[ob + 16 + l16] = (bf16)o1[r];
    }
  }
}

// ---------------------------------------------------------------------------
extern "C" void kernel_launch(void* const* d_in, const int* in_sizes, int n_in,
                              void* d_out, int out_size, void* d_ws, size_t ws_size,
                              hipStream_t stream) {
  const float* x      = (const float*)d_in[0];
  const float* ln1_g  = (const float*)d_in[1];
  const float* ln1_b  = (const float*)d_in[2];
  const float* ln2_g  = (const float*)d_in[3];
  const float* ln2_b  = (const float*)d_in[4];
  const float* qkv_w  = (const float*)d_in[5];
  const float* qkv_b  = (const float*)d_in[6];
  const float* proj_w = (const float*)d_in[7];
  const float* proj_b = (const float*)d_in[8];
  const float* mlp_w1 = (const float*)d_in[9];
  const float* mlp_b1 = (const float*)d_in[10];
  const float* mlp_w2 = (const float*)d_in[11];
  const float* mlp_b2 = (const float*)d_in[12];
  const float* relt   = (const float*)d_in[13];
  float* out = (float*)d_out;

  char* wsb = (char*)d_ws;
  size_t off = 0;
  auto alloc = [&](size_t bytes) -> void* {
    void* p = wsb + off;
    off += (bytes + 255) & ~(size_t)255;
    return p;
  };

  bf16* wTqkv  = (bf16*)alloc((size_t)1152 * 384 * 2);
  bf16* wTproj = (bf16*)alloc((size_t)384 * 384 * 2);
  bf16* wTm1   = (bf16*)alloc((size_t)1536 * 384 * 2);
  bf16* wTm2   = (bf16*)alloc((size_t)384 * 1536 * 2);
  bf16* wins   = (bf16*)alloc((size_t)MTOK * Cc * 2);
  bf16* qb     = (bf16*)alloc((size_t)NW * NHh * N_ * HD * 2);
  bf16* kb     = (bf16*)alloc((size_t)NW * NHh * N_ * HD * 2);
  bf16* vt     = (bf16*)alloc((size_t)NW * NHh * HD * 64 * 2);
  float* x2    = (float*)alloc((size_t)MTOK * Cc * 4);
  bf16* m1     = (bf16*)alloc((size_t)MTOK * 1536 * 2);
  bf16* attn_o = wins;  // reuse: wins consumed by QKV GEMM before attn writes
  bf16* h2     = qb;    // reuse: q consumed by attention before LN2 writes

  // weight prep (transpose + bf16)
  {
    int n;
    n = 384 * 1152; convert_transpose<<<(n + 255) / 256, 256, 0, stream>>>(qkv_w,  wTqkv,  384, 1152);
    n = 384 * 384;  convert_transpose<<<(n + 255) / 256, 256, 0, stream>>>(proj_w, wTproj, 384, 384);
    n = 384 * 1536; convert_transpose<<<(n + 255) / 256, 256, 0, stream>>>(mlp_w1, wTm1,   384, 1536);
    n = 1536 * 384; convert_transpose<<<(n + 255) / 256, 256, 0, stream>>>(mlp_w2, wTm2,  1536, 384);
  }

  // LN1 + shift + window partition
  ln_kernel<true><<<MTOK / 8, 256, 0, stream>>>(x, ln1_g, ln1_b, wins);

  // zero v^T padding (columns 49..63)
  hipMemsetAsync(vt, 0, (size_t)NW * NHh * HD * 64 * 2, stream);

  // QKV GEMM + scatter
  {
    dim3 g(1152 / 64, MTOK / 64);
    gemm_bf16_kernel<0><<<g, 128, 0, stream>>>(wins, wTqkv, qkv_b, 384, 1152,
                                               nullptr, nullptr, nullptr, qb, kb, vt);
  }

  // attention
  attn_kernel<<<NW * NHh, 128, 0, stream>>>(qb, kb, vt, relt, attn_o);

  // proj GEMM + window reverse + un-shift + residual -> x2
  {
    dim3 g(384 / 64, MTOK / 64);
    gemm_bf16_kernel<2><<<g, 128, 0, stream>>>(attn_o, wTproj, proj_b, 384, 384,
                                               nullptr, x2, x, nullptr, nullptr, nullptr);
  }

  // LN2
  ln_kernel<false><<<MTOK / 8, 256, 0, stream>>>(x2, ln2_g, ln2_b, h2);

  // MLP1 + GELU
  {
    dim3 g(1536 / 64, MTOK / 64);
    gemm_bf16_kernel<1><<<g, 128, 0, stream>>>(h2, wTm1, mlp_b1, 384, 1536,
                                               m1, nullptr, nullptr, nullptr, nullptr, nullptr);
  }

  // MLP2 + residual -> out
  {
    dim3 g(384 / 64, MTOK / 64);
    gemm_bf16_kernel<3><<<g, 128, 0, stream>>>(m1, wTm2, mlp_b2, 1536, 384,
                                               nullptr, out, x2, nullptr, nullptr, nullptr);
  }
}
